// IGAT_28819230556368
// MI455X (gfx1250) — matrix operand order repaired and checked by reading, not compiled, not run on hardware
//
#include <hip/hip_runtime.h>
#include <math.h>

#define F_IN   512
#define ATTN_D 64
#define F_OUT  256
#define CG_ITERS 200

typedef __attribute__((ext_vector_type(16))) _Float16    v16h;
typedef __attribute__((ext_vector_type(8)))  float       v8f;
typedef __attribute__((ext_vector_type(4)))  unsigned    u32x4;

union HalfVec { u32x4 u[2]; v16h v; };

__device__ __forceinline__ float waveReduceSum(float s) {
#pragma unroll
    for (int off = 16; off > 0; off >>= 1) s += __shfl_xor(s, off, 32);
    return s;
}

// ---------------------------------------------------------------- utilities
__global__ void zero_f32_kernel(float* p, size_t n) {
    size_t i = (size_t)blockIdx.x * blockDim.x + threadIdx.x;
    size_t st = (size_t)gridDim.x * blockDim.x;
    for (; i < n; i += st) p[i] = 0.0f;
}

__global__ void f32_to_f16_kernel(const float* __restrict__ src,
                                  _Float16* __restrict__ dst, size_t n) {
    size_t i = (size_t)blockIdx.x * blockDim.x + threadIdx.x;
    size_t st = (size_t)gridDim.x * blockDim.x;
    for (; i < n; i += st) dst[i] = (_Float16)src[i];
}

// src: [K, ncol] row-major f32  ->  dst: [ncol, K] row-major f16
__global__ void transpose_f16_kernel(const float* __restrict__ src,
                                     _Float16* __restrict__ dst,
                                     int K, int ncol) {
    size_t total = (size_t)K * ncol;
    size_t i = (size_t)blockIdx.x * blockDim.x + threadIdx.x;
    size_t st = (size_t)gridDim.x * blockDim.x;
    for (; i < total; i += st) {
        int k = (int)(i / ncol);
        int n = (int)(i % ncol);
        dst[(size_t)n * K + k] = (_Float16)src[i];
    }
}

// ---------------------------------------------------------------- WMMA GEMM
// C[M, NCOLS] = A[M, 512](f16) * B[512, NCOLS] + bias ; BT is B transposed
// [NCOLS, 512] f16. One wave -> one 16x16 tile, K stepped by 32.
template <int NCOLS>
__global__ void wmma_gemm_kernel(const _Float16* __restrict__ A,
                                 const _Float16* __restrict__ BT,
                                 const float* __restrict__ bias,
                                 float* __restrict__ C,
                                 int mtiles) {
    const int lane = threadIdx.x & 31;
    const int wv   = threadIdx.x >> 5;
    const int tm   = blockIdx.x * 8 + wv;        // 8 waves / block
    if (tm >= mtiles) return;                    // wave-uniform exit
    const int tn = blockIdx.y;
    const int hi = lane >> 4;
    const int m  = lane & 15;

    const _Float16* arow = A  + (size_t)(tm * 16 + m) * F_IN;
    const _Float16* brow = BT + (size_t)(tn * 16 + m) * F_IN;

    v8f acc = {};
#pragma unroll
    for (int k0 = 0; k0 < F_IN; k0 += 32) {
        HalfVec fa, fb;
        // A 16x32 f16 frag: lanes 0-15 K {k0..k0+7, k0+16..k0+23}; lanes 16-31 shifted by 8
        fa.u[0] = *(const u32x4*)(arow + k0 + hi * 8);
        fa.u[1] = *(const u32x4*)(arow + k0 + 16 + hi * 8);
        // B 32x16 f16 frag: lane = column, same K as A: lanes 0-15 K {k0..k0+7, k0+16..k0+23}; lanes 16-31 shifted by 8
        fb.u[0] = *(const u32x4*)(brow + k0 + hi * 8);
        fb.u[1] = *(const u32x4*)(brow + k0 + 16 + hi * 8);
        acc = __builtin_amdgcn_wmma_f32_16x16x32_f16(
            false, fa.v, false, fb.v, (short)0, acc, false, false);
    }

    const int ncol = tn * 16 + m;
    const float bs = bias[ncol];
    const int mbase = tm * 16 + hi * 8;
#pragma unroll
    for (int r = 0; r < 8; ++r)
        C[(size_t)(mbase + r) * NCOLS + ncol] = acc[r] + bs;
}

// ---------------------------------------------------------------- edge attention
// one wave per edge: vals[e] = sigmoid(dot64(attn[row], attn[col])); deg += vals
__global__ void edge_attn_kernel(const int* __restrict__ row,
                                 const int* __restrict__ col,
                                 const float* __restrict__ attn,
                                 float* __restrict__ vals,
                                 float* __restrict__ deg, int E) {
    size_t gwave = ((size_t)blockIdx.x * blockDim.x + threadIdx.x) >> 5;
    int lane = threadIdx.x & 31;
    if (gwave >= (size_t)E) return;
    int r = row[gwave], c = col[gwave];
    const float* ar = attn + (size_t)r * ATTN_D;
    const float* ac = attn + (size_t)c * ATTN_D;
    float s = ar[lane] * ac[lane] + ar[lane + 32] * ac[lane + 32];
    s = waveReduceSum(s);
    if (lane == 0) {
        float v = 1.0f / (1.0f + expf(-s));
        vals[gwave] = v;
        atomicAdd(deg + r, v);
    }
}

__global__ void dinv_kernel(const float* __restrict__ deg,
                            float* __restrict__ dinv, int n) {
    int i = blockIdx.x * blockDim.x + threadIdx.x;
    if (i < n) { float d = deg[i]; dinv[i] = (d > 0.0f) ? rsqrtf(d) : 0.0f; }
}

// vals[e] *= coef * dinv[row]*dinv[col]   (coef = 1 - sigmoid(eps_inv))
__global__ void edge_norm_kernel(float* __restrict__ vals,
                                 const float* __restrict__ dinv,
                                 const int* __restrict__ row,
                                 const int* __restrict__ col,
                                 const float* __restrict__ eps_inv, int E) {
    int e = blockIdx.x * blockDim.x + threadIdx.x;
    if (e >= E) return;
    float eps  = 1.0f / (1.0f + expf(-eps_inv[0]));
    float coef = 1.0f - eps;
    vals[e] = coef * vals[e] * dinv[row[e]] * dinv[col[e]];
}

// ---------------------------------------------------------------- CSR build
__global__ void count_rows_kernel(const int* __restrict__ row,
                                  int* __restrict__ cnt, int E) {
    int e = blockIdx.x * blockDim.x + threadIdx.x;
    if (e < E) atomicAdd(cnt + row[e], 1);
}

// single-block exclusive scan over n<=50176 ints
__global__ void scan_kernel(const int* __restrict__ cnt,
                            int* __restrict__ rowptr, int n) {
    __shared__ int partial[1024];
    int t = threadIdx.x;
    int chunk = (n + 1023) / 1024;
    int start = t * chunk;
    int end   = start + chunk; if (end > n) end = n;
    int s = 0;
    for (int i = start; i < end; ++i) s += cnt[i];
    partial[t] = s;
    __syncthreads();
    for (int off = 1; off < 1024; off <<= 1) {
        int v = (t >= off) ? partial[t - off] : 0;
        __syncthreads();
        if (t >= off) partial[t] += v;
        __syncthreads();
    }
    int excl = (t == 0) ? 0 : partial[t - 1];
    for (int i = start; i < end; ++i) { rowptr[i] = excl; excl += cnt[i]; }
    if (t == 1023) rowptr[n] = partial[1023];
}

__global__ void build_csr_kernel(const int* __restrict__ row,
                                 const int* __restrict__ col,
                                 const float* __restrict__ vals,
                                 const int* __restrict__ rowptr,
                                 int* __restrict__ fill,
                                 int* __restrict__ colIdx,
                                 float* __restrict__ cw, int E) {
    int e = blockIdx.x * blockDim.x + threadIdx.x;
    if (e >= E) return;
    int r = row[e];
    int pos = rowptr[r] + atomicAdd(fill + r, 1);
    colIdx[pos] = col[e];
    cw[pos]     = vals[e];
}

// ---------------------------------------------------------------- CG kernels
// Ap = p - sum_{j in row i} cw[j] * p[colIdx[j]] ; block = row, lane = feature
__global__ void lmatvec_kernel(const int* __restrict__ rowptr,
                               const int* __restrict__ colIdx,
                               const float* __restrict__ cw,
                               const float* __restrict__ p,
                               float* __restrict__ Ap) {
    int i = blockIdx.x;
    int f = threadIdx.x;
    float acc = p[(size_t)i * F_OUT + f];
    int s = rowptr[i], e = rowptr[i + 1];
    for (int j = s; j < e; ++j)
        acc -= cw[j] * p[(size_t)colIdx[j] * F_OUT + f];
    Ap[(size_t)i * F_OUT + f] = acc;
}

__global__ void copy_rp_kernel(const float* __restrict__ b,
                               float* __restrict__ r,
                               float* __restrict__ p, size_t n) {
    size_t i = (size_t)blockIdx.x * blockDim.x + threadIdx.x;
    size_t st = (size_t)gridDim.x * blockDim.x;
    for (; i < n; i += st) { float v = b[i]; r[i] = v; p[i] = v; }
}

__global__ void dot_kernel(const float* __restrict__ a,
                           const float* __restrict__ b,
                           float* __restrict__ out, size_t n) {
    size_t i = (size_t)blockIdx.x * blockDim.x + threadIdx.x;
    size_t st = (size_t)gridDim.x * blockDim.x;
    float acc = 0.0f;
    for (; i < n; i += st) acc += a[i] * b[i];
    acc = waveReduceSum(acc);
    if ((threadIdx.x & 31) == 0) atomicAdd(out, acc);
}

// S: [0]=rs_old [1]=pAp [2]=rs_new [3]=alpha [4]=beta
__global__ void cg_alpha_kernel(float* S) {
    S[3] = S[0] / fmaxf(S[1], 1e-30f);   // alpha -> 0 once converged
    S[2] = 0.0f;
}

__global__ void axpy_rr_kernel(float* __restrict__ x,
                               float* __restrict__ r,
                               const float* __restrict__ p,
                               const float* __restrict__ Ap,
                               float* __restrict__ S, size_t n) {
    float alpha = S[3];
    size_t i = (size_t)blockIdx.x * blockDim.x + threadIdx.x;
    size_t st = (size_t)gridDim.x * blockDim.x;
    float acc = 0.0f;
    for (; i < n; i += st) {
        x[i] += alpha * p[i];
        float ri = r[i] - alpha * Ap[i];
        r[i] = ri;
        acc += ri * ri;
    }
    acc = waveReduceSum(acc);
    if ((threadIdx.x & 31) == 0) atomicAdd(S + 2, acc);
}

__global__ void cg_beta_kernel(float* S) {
    S[4] = S[2] / fmaxf(S[0], 1e-30f);
    S[0] = S[2];
    S[1] = 0.0f;
}

__global__ void update_p_kernel(float* __restrict__ p,
                                const float* __restrict__ r,
                                const float* __restrict__ S, size_t n) {
    float beta = S[4];
    size_t i = (size_t)blockIdx.x * blockDim.x + threadIdx.x;
    size_t st = (size_t)gridDim.x * blockDim.x;
    for (; i < n; i += st) p[i] = r[i] + beta * p[i];
}

// ---------------------------------------------------------------- launch
static inline char* carve(char*& p, size_t bytes) {
    char* r = p;
    p += (bytes + 255) & ~(size_t)255;
    return r;
}

extern "C" void kernel_launch(void* const* d_in, const int* in_sizes, int n_in,
                              void* d_out, int out_size, void* d_ws, size_t ws_size,
                              hipStream_t stream) {
    const int*   row     = (const int*)d_in[0];
    const int*   col     = (const int*)d_in[1];
    const float* feat    = (const float*)d_in[2];
    const float* Wa      = (const float*)d_in[3];
    const float* ba      = (const float*)d_in[4];
    const float* Wo      = (const float*)d_in[5];
    const float* bo      = (const float*)d_in[6];
    const float* eps_inv = (const float*)d_in[7];

    const int E = in_sizes[0];
    const int N = in_sizes[2] / F_IN;
    const size_t NF = (size_t)N * F_OUT;

    char* w = (char*)d_ws;
    _Float16* feat_h = (_Float16*)carve(w, (size_t)N * F_IN * sizeof(_Float16));
    _Float16* WaT    = (_Float16*)carve(w, (size_t)ATTN_D * F_IN * sizeof(_Float16));
    _Float16* WoT    = (_Float16*)carve(w, (size_t)F_OUT * F_IN * sizeof(_Float16));
    float* attn      = (float*)carve(w, (size_t)N * ATTN_D * sizeof(float));
    float* vals      = (float*)carve(w, (size_t)E * sizeof(float));
    float* deg       = (float*)carve(w, (size_t)N * sizeof(float));
    float* dinv      = (float*)carve(w, (size_t)N * sizeof(float));
    int*   rcnt      = (int*)  carve(w, (size_t)N * sizeof(int));
    int*   rowptr    = (int*)  carve(w, (size_t)(N + 1) * sizeof(int));
    int*   fill      = (int*)  carve(w, (size_t)N * sizeof(int));
    int*   colIdx    = (int*)  carve(w, (size_t)E * sizeof(int));
    float* cw        = (float*)carve(w, (size_t)E * sizeof(float));
    float* bvec      = (float*)carve(w, NF * sizeof(float));
    float* rvec      = (float*)carve(w, NF * sizeof(float));
    float* pvec      = (float*)carve(w, NF * sizeof(float));
    float* Ap        = (float*)carve(w, NF * sizeof(float));
    float* S         = (float*)carve(w, 8 * sizeof(float));
    float* x         = (float*)d_out;

    const int GS = 2048;           // grid-stride blocks
    // ---- zero init
    zero_f32_kernel<<<GS, 256, 0, stream>>>((float*)rcnt, (size_t)N);
    zero_f32_kernel<<<GS, 256, 0, stream>>>((float*)fill, (size_t)N);
    zero_f32_kernel<<<GS, 256, 0, stream>>>(deg, (size_t)N);
    zero_f32_kernel<<<1, 32, 0, stream>>>(S, 8);
    zero_f32_kernel<<<GS, 256, 0, stream>>>(x, NF);

    // ---- f16 staging for WMMA
    f32_to_f16_kernel<<<GS, 256, 0, stream>>>(feat, feat_h, (size_t)N * F_IN);
    transpose_f16_kernel<<<256, 256, 0, stream>>>(Wa, WaT, F_IN, ATTN_D);
    transpose_f16_kernel<<<512, 256, 0, stream>>>(Wo, WoT, F_IN, F_OUT);

    // ---- WMMA GEMMs: attn = feat@Wa + ba ; b = feat@Wo + bo
    const int mtiles = (N + 15) / 16;                    // 3125
    dim3 gblk((mtiles + 7) / 8, ATTN_D / 16);
    wmma_gemm_kernel<ATTN_D><<<gblk, 256, 0, stream>>>(feat_h, WaT, ba, attn, mtiles);
    dim3 gblk2((mtiles + 7) / 8, F_OUT / 16);
    wmma_gemm_kernel<F_OUT><<<gblk2, 256, 0, stream>>>(feat_h, WoT, bo, bvec, mtiles);

    // ---- edge attention + degree
    size_t ethreads = (size_t)E * 32;
    edge_attn_kernel<<<(unsigned)((ethreads + 255) / 256), 256, 0, stream>>>(
        row, col, attn, vals, deg, E);
    dinv_kernel<<<(N + 255) / 256, 256, 0, stream>>>(deg, dinv, N);
    edge_norm_kernel<<<(E + 255) / 256, 256, 0, stream>>>(vals, dinv, row, col, eps_inv, E);

    // ---- CSR build
    count_rows_kernel<<<(E + 255) / 256, 256, 0, stream>>>(row, rcnt, E);
    scan_kernel<<<1, 1024, 0, stream>>>(rcnt, rowptr, N);
    build_csr_kernel<<<(E + 255) / 256, 256, 0, stream>>>(
        row, col, vals, rowptr, fill, colIdx, cw, E);

    // ---- CG init: x=0, r=p=b, S[0]=<r,r>
    copy_rp_kernel<<<GS, 256, 0, stream>>>(bvec, rvec, pvec, NF);
    dot_kernel<<<GS, 256, 0, stream>>>(rvec, rvec, S + 0, NF);

    // ---- CG iterations (fixed count; alpha self-freezes at convergence)
    for (int it = 0; it < CG_ITERS; ++it) {
        lmatvec_kernel<<<N, F_OUT, 0, stream>>>(rowptr, colIdx, cw, pvec, Ap);
        dot_kernel<<<GS, 256, 0, stream>>>(pvec, Ap, S + 1, NF);
        cg_alpha_kernel<<<1, 1, 0, stream>>>(S);
        axpy_rr_kernel<<<GS, 256, 0, stream>>>(x, rvec, pvec, Ap, S, NF);
        cg_beta_kernel<<<1, 1, 0, stream>>>(S);
        update_p_kernel<<<GS, 256, 0, stream>>>(pvec, rvec, S, NF);
    }
}
